// MoELayer_86689619903476
// MI455X (gfx1250) — matrix operand order repaired and checked by reading, not compiled, not run on hardware
//
#include <hip/hip_runtime.h>
#include <hip/hip_bf16.h>

typedef __attribute__((ext_vector_type(16))) __bf16 bf16x16;
typedef __attribute__((ext_vector_type(8)))  float  f32x8;

union FragBF { uint4 u[2]; bf16x16 v; };

#define NTOK 4096
#define DDIM 1024
#define HDIM 256
#define EEXP 8
#define FDIM 1408

#define BM 128
#define BN 64
#define BK 32
#define LDA_S 40   // padded LDS row stride (elements) for A tile
#define LDB_S 40   // padded LDS row stride (elements) for B tile

// ---- CDNA5 async global->LDS staging (ASYNCcnt path), with sync fallback ----
#if defined(__has_builtin)
# if __has_builtin(__builtin_amdgcn_global_load_async_to_lds_b128)
#  define ASYNC_LDS 1
# endif
#endif

#ifdef ASYNC_LDS
typedef int gcc_v4i __attribute__((__vector_size__(4 * sizeof(int))));
typedef __attribute__((address_space(1))) gcc_v4i as1_v4i;
typedef __attribute__((address_space(3))) gcc_v4i as3_v4i;
static __device__ __forceinline__ void async_cp16(const unsigned short* g, unsigned short* l) {
  __builtin_amdgcn_global_load_async_to_lds_b128((as1_v4i*)g, (as3_v4i*)l, 0, 0);
}
#define STAGE16(g, l) async_cp16((g), (l))
// wait until at most n async ops outstanding (older tile group guaranteed landed)
#define ASYNC_WAIT(n) asm volatile("s_wait_asynccnt %0" :: "n"(n) : "memory")
#else
#define STAGE16(g, l) (*(uint4*)(l) = *(const uint4*)(g))
#define ASYNC_WAIT(n) ((void)0)
#endif

static __device__ __forceinline__ unsigned short f2bf(float f) {
  unsigned u = __float_as_uint(f);
  u += 0x7FFFu + ((u >> 16) & 1u);     // round-to-nearest-even
  return (unsigned short)(u >> 16);
}
static __device__ __forceinline__ float bf2f(unsigned short b) {
  return __uint_as_float(((unsigned)b) << 16);
}

// ---------------- elementwise fp32 -> bf16 ----------------
__global__ __launch_bounds__(256)
void k_cvt(const float* __restrict__ in, unsigned short* __restrict__ out, int n) {
  int i = blockIdx.x * 256 + threadIdx.x;
  if (i < n) out[i] = f2bf(in[i]);
}

// -------- batched transpose + convert: in [b][R][C] f32 -> out [b][C][R] bf16 --------
__global__ __launch_bounds__(256)
void k_tcvt(const float* __restrict__ in, unsigned short* __restrict__ out, int R, int C) {
  int i = blockIdx.x * 256 + threadIdx.x;
  int b = blockIdx.y;
  if (i >= R * C) return;
  int r = i / C, c = i % C;
  size_t base = (size_t)b * R * C;
  out[base + (size_t)c * R + r] = f2bf(in[base + i]);
}

// =====================================================================
//  WMMA GEMM 1: Cb = relu(A @ B^T + bias) -> bf16   (router hidden)
//  A: [M][K] bf16, Bt: [Nc][K] bf16.  K % 64 == 0.
// =====================================================================
__global__ __launch_bounds__(256)
void k_gemm_hidden(const unsigned short* __restrict__ A,
                   const unsigned short* __restrict__ Bt,
                   const float* __restrict__ bias,
                   unsigned short* __restrict__ Cb,
                   int M, int Nc, int K)
{
  __shared__ __align__(16) unsigned short lA[2][BM * LDA_S];
  __shared__ __align__(16) unsigned short lB[2][BN * LDB_S];
  int t = threadIdx.x;
  int mBase = blockIdx.y * BM, nBase = blockIdx.x * BN;
  int wave = t >> 5, lane = t & 31;
  int wm0 = (wave >> 1) * 32, wn0 = (wave & 1) * 32;
  int lr = lane & 15, lh = lane >> 4;
  int am = t >> 1, ak = (t & 1) * 16;
  int bn = t >> 2, bk = (t & 3) * 8;

  f32x8 acc[2][2] = {};

  auto issueT = [&](int buf, int kt) {
    const unsigned short* gA0 = A + (size_t)(mBase + am) * K + kt + ak;
    STAGE16(gA0,     &lA[buf][am * LDA_S + ak]);
    STAGE16(gA0 + 8, &lA[buf][am * LDA_S + ak + 8]);
    STAGE16(Bt + (size_t)(nBase + bn) * K + kt + bk, &lB[buf][bn * LDB_S + bk]);
  };
  auto comp = [&](int buf) {
    FragBF fa[2], fb[2];
#pragma unroll
    for (int i = 0; i < 2; i++) {
      int row = wm0 + i * 16 + lr;
      fa[i].u[0] = *(const uint4*)&lA[buf][row * LDA_S + lh * 8];
      fa[i].u[1] = *(const uint4*)&lA[buf][row * LDA_S + 16 + lh * 8];
    }
#pragma unroll
    for (int j = 0; j < 2; j++) {
      int col = wn0 + j * 16 + lr;
      fb[j].u[0] = *(const uint4*)&lB[buf][col * LDB_S + lh * 8];
      fb[j].u[1] = *(const uint4*)&lB[buf][col * LDB_S + 16 + lh * 8];
    }
#pragma unroll
    for (int i = 0; i < 2; i++)
#pragma unroll
      for (int j = 0; j < 2; j++)
        acc[i][j] = __builtin_amdgcn_wmma_f32_16x16x32_bf16(
            false, fa[i].v, false, fb[j].v, (short)0, acc[i][j], false, false);
  };

  issueT(0, 0);
  for (int kt = 0; kt < K; kt += 2 * BK) {
    issueT(1, kt + BK);
    ASYNC_WAIT(3);
    __syncthreads();
    comp(0);
    __syncthreads();
    if (kt + 2 * BK < K) { issueT(0, kt + 2 * BK); ASYNC_WAIT(3); }
    else                 { ASYNC_WAIT(0); }
    __syncthreads();
    comp(1);
    __syncthreads();
  }
#pragma unroll
  for (int i = 0; i < 2; i++)
#pragma unroll
    for (int j = 0; j < 2; j++) {
      int col = nBase + wn0 + j * 16 + lr;
      float bc = bias[col];
#pragma unroll
      for (int v = 0; v < 8; v++) {
        int row = mBase + wm0 + i * 16 + v + 8 * lh;
        float x = acc[i][j][v] + bc;
        Cb[(size_t)row * Nc + col] = f2bf(fmaxf(x, 0.f));
      }
    }
}

// =====================================================================
//  WMMA GEMM 2 (dual-B): GU = r[:,e] * silu(A@Bg^T) * (A@Bu^T) -> bf16
// =====================================================================
__global__ __launch_bounds__(256)
void k_gemm_gu(const unsigned short* __restrict__ A,
               const unsigned short* __restrict__ Bg,
               const unsigned short* __restrict__ Bu,
               const float* __restrict__ rW, int e,
               unsigned short* __restrict__ GU,
               int M, int Nc, int K)
{
  __shared__ __align__(16) unsigned short lA [2][BM * LDA_S];
  __shared__ __align__(16) unsigned short lBg[2][BN * LDB_S];
  __shared__ __align__(16) unsigned short lBu[2][BN * LDB_S];
  __shared__ float rs[BM];
  int t = threadIdx.x;
  int mBase = blockIdx.y * BM, nBase = blockIdx.x * BN;
  int wave = t >> 5, lane = t & 31;
  int wm0 = (wave >> 1) * 32, wn0 = (wave & 1) * 32;
  int lr = lane & 15, lh = lane >> 4;
  int am = t >> 1, ak = (t & 1) * 16;
  int bn = t >> 2, bk = (t & 3) * 8;

  if (t < BM) rs[t] = rW[(size_t)(mBase + t) * EEXP + e];

  f32x8 aG[2][2] = {}, aU[2][2] = {};

  auto issueT = [&](int buf, int kt) {
    const unsigned short* gA0 = A + (size_t)(mBase + am) * K + kt + ak;
    STAGE16(gA0,     &lA[buf][am * LDA_S + ak]);
    STAGE16(gA0 + 8, &lA[buf][am * LDA_S + ak + 8]);
    STAGE16(Bg + (size_t)(nBase + bn) * K + kt + bk, &lBg[buf][bn * LDB_S + bk]);
    STAGE16(Bu + (size_t)(nBase + bn) * K + kt + bk, &lBu[buf][bn * LDB_S + bk]);
  };
  auto comp = [&](int buf) {
    FragBF fa[2], fg[2], fu[2];
#pragma unroll
    for (int i = 0; i < 2; i++) {
      int row = wm0 + i * 16 + lr;
      fa[i].u[0] = *(const uint4*)&lA[buf][row * LDA_S + lh * 8];
      fa[i].u[1] = *(const uint4*)&lA[buf][row * LDA_S + 16 + lh * 8];
    }
#pragma unroll
    for (int j = 0; j < 2; j++) {
      int col = wn0 + j * 16 + lr;
      fg[j].u[0] = *(const uint4*)&lBg[buf][col * LDB_S + lh * 8];
      fg[j].u[1] = *(const uint4*)&lBg[buf][col * LDB_S + 16 + lh * 8];
      fu[j].u[0] = *(const uint4*)&lBu[buf][col * LDB_S + lh * 8];
      fu[j].u[1] = *(const uint4*)&lBu[buf][col * LDB_S + 16 + lh * 8];
    }
#pragma unroll
    for (int i = 0; i < 2; i++)
#pragma unroll
      for (int j = 0; j < 2; j++) {
        aG[i][j] = __builtin_amdgcn_wmma_f32_16x16x32_bf16(
            false, fa[i].v, false, fg[j].v, (short)0, aG[i][j], false, false);
        aU[i][j] = __builtin_amdgcn_wmma_f32_16x16x32_bf16(
            false, fa[i].v, false, fu[j].v, (short)0, aU[i][j], false, false);
      }
  };

  issueT(0, 0);
  for (int kt = 0; kt < K; kt += 2 * BK) {
    issueT(1, kt + BK);
    ASYNC_WAIT(4);
    __syncthreads();
    comp(0);
    __syncthreads();
    if (kt + 2 * BK < K) { issueT(0, kt + 2 * BK); ASYNC_WAIT(4); }
    else                 { ASYNC_WAIT(0); }
    __syncthreads();
    comp(1);
    __syncthreads();
  }
#pragma unroll
  for (int i = 0; i < 2; i++)
#pragma unroll
    for (int j = 0; j < 2; j++) {
      int col = nBase + wn0 + j * 16 + lr;
#pragma unroll
      for (int v = 0; v < 8; v++) {
        int rl = wm0 + i * 16 + v + 8 * lh;
        float g = aG[i][j][v], u = aU[i][j][v];
        float s = g / (1.f + __expf(-g));          // silu
        GU[(size_t)(mBase + rl) * Nc + col] = f2bf(rs[rl] * s * u);
      }
    }
}

// =====================================================================
//  WMMA GEMM 3: Y (+)= A @ B^T (+ bias) -> fp32   (down-proj / decoder)
// =====================================================================
__global__ __launch_bounds__(256)
void k_gemm_down(const unsigned short* __restrict__ A,
                 const unsigned short* __restrict__ Bt,
                 const float* __restrict__ bias,   // nullable
                 float* __restrict__ Y,
                 int accum, int M, int Nc, int K)
{
  __shared__ __align__(16) unsigned short lA[2][BM * LDA_S];
  __shared__ __align__(16) unsigned short lB[2][BN * LDB_S];
  int t = threadIdx.x;
  int mBase = blockIdx.y * BM, nBase = blockIdx.x * BN;
  int wave = t >> 5, lane = t & 31;
  int wm0 = (wave >> 1) * 32, wn0 = (wave & 1) * 32;
  int lr = lane & 15, lh = lane >> 4;
  int am = t >> 1, ak = (t & 1) * 16;
  int bn = t >> 2, bk = (t & 3) * 8;

  f32x8 acc[2][2] = {};

  auto issueT = [&](int buf, int kt) {
    const unsigned short* gA0 = A + (size_t)(mBase + am) * K + kt + ak;
    STAGE16(gA0,     &lA[buf][am * LDA_S + ak]);
    STAGE16(gA0 + 8, &lA[buf][am * LDA_S + ak + 8]);
    STAGE16(Bt + (size_t)(nBase + bn) * K + kt + bk, &lB[buf][bn * LDB_S + bk]);
  };
  auto comp = [&](int buf) {
    FragBF fa[2], fb[2];
#pragma unroll
    for (int i = 0; i < 2; i++) {
      int row = wm0 + i * 16 + lr;
      fa[i].u[0] = *(const uint4*)&lA[buf][row * LDA_S + lh * 8];
      fa[i].u[1] = *(const uint4*)&lA[buf][row * LDA_S + 16 + lh * 8];
    }
#pragma unroll
    for (int j = 0; j < 2; j++) {
      int col = wn0 + j * 16 + lr;
      fb[j].u[0] = *(const uint4*)&lB[buf][col * LDB_S + lh * 8];
      fb[j].u[1] = *(const uint4*)&lB[buf][col * LDB_S + 16 + lh * 8];
    }
#pragma unroll
    for (int i = 0; i < 2; i++)
#pragma unroll
      for (int j = 0; j < 2; j++)
        acc[i][j] = __builtin_amdgcn_wmma_f32_16x16x32_bf16(
            false, fa[i].v, false, fb[j].v, (short)0, acc[i][j], false, false);
  };

  issueT(0, 0);
  for (int kt = 0; kt < K; kt += 2 * BK) {
    issueT(1, kt + BK);
    ASYNC_WAIT(3);
    __syncthreads();
    comp(0);
    __syncthreads();
    if (kt + 2 * BK < K) { issueT(0, kt + 2 * BK); ASYNC_WAIT(3); }
    else                 { ASYNC_WAIT(0); }
    __syncthreads();
    comp(1);
    __syncthreads();
  }
#pragma unroll
  for (int i = 0; i < 2; i++)
#pragma unroll
    for (int j = 0; j < 2; j++) {
      int col = nBase + wn0 + j * 16 + lr;
      float bc = bias ? bias[col] : 0.f;
#pragma unroll
      for (int v = 0; v < 8; v++) {
        int row = mBase + wm0 + i * 16 + v + 8 * lh;
        size_t idx = (size_t)row * Nc + col;
        float val = acc[i][j][v] + bc;
        if (accum) Y[idx] += val; else Y[idx] = val;
      }
    }
}

// ---------------- router tail: z, softmax, alpha_q, r, alpha_p ----------------
__global__ __launch_bounds__(256)
void k_router(const unsigned short* __restrict__ Hb,   // [N][H] bf16 (already relu'd)
              const float* __restrict__ Wr2, const float* __restrict__ br2,
              float* __restrict__ zO, float* __restrict__ apO,
              float* __restrict__ aqO, float* __restrict__ rO)
{
  int n = blockIdx.x * 256 + threadIdx.x;
  if (n >= NTOK) return;
  float z[EEXP];
#pragma unroll
  for (int e = 0; e < EEXP; e++) z[e] = br2[e];
  for (int h = 0; h < HDIM; h++) {
    float hv = bf2f(Hb[(size_t)n * HDIM + h]);
#pragma unroll
    for (int e = 0; e < EEXP; e++) z[e] += hv * Wr2[h * EEXP + e];
  }
  float m = z[0];
#pragma unroll
  for (int e = 1; e < EEXP; e++) m = fmaxf(m, z[e]);
  float p[EEXP], s = 0.f;
#pragma unroll
  for (int e = 0; e < EEXP; e++) { p[e] = __expf(z[e] - m); s += p[e]; }
  float aq[EEXP], aqs = 0.f;
#pragma unroll
  for (int e = 0; e < EEXP; e++) { aq[e] = 1e-3f + 20.f * (p[e] / s); aqs += aq[e]; }
#pragma unroll
  for (int e = 0; e < EEXP; e++) {
    size_t i = (size_t)n * EEXP + e;
    zO[i]  = z[e];
    aqO[i] = aq[e];
    rO[i]  = aq[e] / aqs;
    float th = 1.f / (1.f + __expf(-0.5f * z[e]));        // sigmoid(z / TAU_Z)
    apO[i] = 0.005f + 1.98f * th;                         // ALPHA_LO + (HI-LO)*theta
  }
}

// ---------------- decoder hidden: Hd = relu(r @ Wd1 + bd1) -> bf16 ----------------
__global__ __launch_bounds__(256)
void k_hd(const float* __restrict__ rW, const float* __restrict__ Wd1,
          const float* __restrict__ bd1, unsigned short* __restrict__ Hd)
{
  int i = blockIdx.x * 256 + threadIdx.x;
  if (i >= NTOK * HDIM) return;
  int n = i >> 8, h = i & 255;
  float a = bd1[h];
#pragma unroll
  for (int e = 0; e < EEXP; e++) a += rW[(size_t)n * EEXP + e] * Wd1[e * HDIM + h];
  Hd[i] = f2bf(fmaxf(a, 0.f));
}

// =====================================================================
extern "C" void kernel_launch(void* const* d_in, const int* in_sizes, int n_in,
                              void* d_out, int out_size, void* d_ws, size_t ws_size,
                              hipStream_t stream) {
  const float* x   = (const float*)d_in[0];
  const float* Wr1 = (const float*)d_in[1];
  const float* br1 = (const float*)d_in[2];
  const float* Wr2 = (const float*)d_in[3];
  const float* br2 = (const float*)d_in[4];
  const float* Wg  = (const float*)d_in[5];
  const float* Wu  = (const float*)d_in[6];
  const float* Wd  = (const float*)d_in[7];
  const float* Wd1 = (const float*)d_in[8];
  const float* bd1 = (const float*)d_in[9];
  const float* Wd2 = (const float*)d_in[10];
  const float* bd2 = (const float*)d_in[11];

  float* yO  = (float*)d_out;                         // (N, D)
  float* zO  = yO + (size_t)NTOK * DDIM;              // (N, E)
  float* apO = zO + (size_t)NTOK * EEXP;              // (N, E)
  float* aqO = apO + (size_t)NTOK * EEXP;             // (N, E)
  float* xrO = aqO + (size_t)NTOK * EEXP;             // (N, D)

  // workspace layout (bf16 stored as u16); all sizes already 256B multiples
  char* ws = (char*)d_ws;
  size_t off = 0;
  auto take = [&](size_t bytes) { char* p = ws + off; off += (bytes + 255) & ~(size_t)255; return p; };
  unsigned short* xb   = (unsigned short*)take((size_t)NTOK * DDIM * 2);
  unsigned short* Wr1t = (unsigned short*)take((size_t)HDIM * DDIM * 2);
  unsigned short* Wgt  = (unsigned short*)take((size_t)EEXP * FDIM * DDIM * 2);
  unsigned short* Wut  = (unsigned short*)take((size_t)EEXP * FDIM * DDIM * 2);
  unsigned short* Wdt  = (unsigned short*)take((size_t)EEXP * DDIM * FDIM * 2);
  unsigned short* Wd2t = (unsigned short*)take((size_t)DDIM * HDIM * 2);
  unsigned short* Hb   = (unsigned short*)take((size_t)NTOK * HDIM * 2);
  unsigned short* Hd   = (unsigned short*)take((size_t)NTOK * HDIM * 2);
  float*          rW   = (float*)take((size_t)NTOK * EEXP * 4);
  unsigned short* GU   = (unsigned short*)take((size_t)NTOK * FDIM * 2);
  (void)ws_size; (void)in_sizes; (void)n_in; (void)out_size;

  // 1) convert / transpose to bf16
  k_cvt<<<(NTOK * DDIM) / 256, 256, 0, stream>>>(x, xb, NTOK * DDIM);
  k_tcvt<<<dim3((DDIM * HDIM) / 256, 1), 256, 0, stream>>>(Wr1, Wr1t, DDIM, HDIM);
  k_tcvt<<<dim3((DDIM * FDIM) / 256, EEXP), 256, 0, stream>>>(Wg, Wgt, DDIM, FDIM);
  k_tcvt<<<dim3((DDIM * FDIM) / 256, EEXP), 256, 0, stream>>>(Wu, Wut, DDIM, FDIM);
  k_tcvt<<<dim3((FDIM * DDIM) / 256, EEXP), 256, 0, stream>>>(Wd, Wdt, FDIM, DDIM);
  k_tcvt<<<dim3((HDIM * DDIM) / 256, 1), 256, 0, stream>>>(Wd2, Wd2t, HDIM, DDIM);

  // 2) router hidden (WMMA), then router tail
  k_gemm_hidden<<<dim3(HDIM / BN, NTOK / BM), 256, 0, stream>>>(
      xb, Wr1t, br1, Hb, NTOK, HDIM, DDIM);
  k_router<<<NTOK / 256, 256, 0, stream>>>(Hb, Wr2, br2, zO, apO, aqO, rW);

  // 3) experts: dual-B gate/up (WMMA) then accumulating down-proj (WMMA)
  for (int e = 0; e < EEXP; e++) {
    k_gemm_gu<<<dim3(FDIM / BN, NTOK / BM), 256, 0, stream>>>(
        xb, Wgt + (size_t)e * FDIM * DDIM, Wut + (size_t)e * FDIM * DDIM,
        rW, e, GU, NTOK, FDIM, DDIM);
    k_gemm_down<<<dim3(DDIM / BN, NTOK / BM), 256, 0, stream>>>(
        GU, Wdt + (size_t)e * DDIM * FDIM, nullptr, yO, (e > 0) ? 1 : 0,
        NTOK, DDIM, FDIM);
  }

  // 4) decoder reconstruction (WMMA)
  k_hd<<<(NTOK * HDIM) / 256, 256, 0, stream>>>(rW, Wd1, bd1, Hd);
  k_gemm_down<<<dim3(DDIM / BN, NTOK / BM), 256, 0, stream>>>(
      Hd, Wd2t, bd2, xrO, 0, NTOK, DDIM, HDIM);
}
